// WindowedGQA_91319594648396
// MI455X (gfx1250) — matrix operand order repaired and checked by reading, not compiled, not run on hardware
//
#include <hip/hip_runtime.h>

// ---------------------------------------------------------------------------
// WindowedGQA on MI455X (gfx1250). All heavy math via v_wmma_f32_16x16x32_f16.
// - ping-pong LDS tiles, 1 barrier per GEMM k-step
// - compile-time N/K so all staging offsets fold into 24-bit immediates
// - single shared address register per staging stream (+imm offsets)
// - global_prefetch_b8 hints for the next-next tile of each stream
// - softmax row pass parallelized 4 threads/row, mask via unsigned-range trick
// ---------------------------------------------------------------------------

typedef __attribute__((ext_vector_type(16))) _Float16       v16h;
typedef __attribute__((ext_vector_type(16))) unsigned short v16u;
typedef __attribute__((ext_vector_type(8)))  float          v8f;
typedef __attribute__((ext_vector_type(4)))  unsigned int   v4u;
typedef __attribute__((ext_vector_type(4)))  float          v4f;

#define B_DIM  2
#define L_DIM  2048
#define D_DIM  2048
#define H_DIM  32
#define HK_DIM 8
#define HD_DIM 64
#define KVD    (HK_DIM * HD_DIM)   // 512
#define WIN    1024

#define AS_STRIDE 40   // 128x32 f16 tile rows, 80B
#define BT_STRIDE 40   // transposed B tile cols, 80B
#define KS_STRIDE 72   // 64x64 f16 tile rows, 144B (cols 64..71 = scratch)
#define SS_STRIDE 68   // 64x64 f32 score tile, 272B (cols 64..67 = scratch)

union FragU {
  v16u f;
  v4u  q[2];
  unsigned int d[8];
};

__device__ __forceinline__ unsigned short f2h(float f) {
  _Float16 h = (_Float16)f;
  return __builtin_bit_cast(unsigned short, h);
}

__device__ __forceinline__ unsigned int pk2h(float x, float y) {
  return (unsigned int)f2h(x) | ((unsigned int)f2h(y) << 16);
}

__device__ __forceinline__ v8f vzero8() {
  v8f z;
#pragma unroll
  for (int i = 0; i < 8; ++i) z[i] = 0.0f;
  return z;
}

__device__ __forceinline__ v8f wmma16(v16u a, v16u b, v8f c) {
  return __builtin_amdgcn_wmma_f32_16x16x32_f16(
      false, __builtin_bit_cast(v16h, a),
      false, __builtin_bit_cast(v16h, b),
      (short)0, c, false, false);
}

// ---------------------------------------------------------------------------
__global__ __launch_bounds__(256) void cvt_f32_to_f16(const float* __restrict__ in,
                                                      unsigned short* __restrict__ out,
                                                      int n8) {
  int i = blockIdx.x * 256 + threadIdx.x;
  if (i < n8) {
    v4f a = *(const v4f*)(in + (size_t)i * 8);
    v4f b = *(const v4f*)(in + (size_t)i * 8 + 4);
    v4u o;
    o[0] = pk2h(a[0], a[1]);
    o[1] = pk2h(a[2], a[3]);
    o[2] = pk2h(b[0], b[1]);
    o[3] = pk2h(b[2], b[3]);
    *(v4u*)(out + (size_t)i * 8) = o;
  }
}

// ---------------------------------------------------------------------------
// f16 GEMM: C[M,N] = A[M,K] * B[K,N], N/K compile-time. Block 128x128, BK=32,
// 8 waves, wave = 64x32 (4x2 accums). Ping-pong LDS, 1 barrier / k-step.
// ---------------------------------------------------------------------------
template <bool STORE_F16, int N, int K>
__global__ __launch_bounds__(256) void gemm_f16_wmma(const unsigned short* __restrict__ A,
                                                     const unsigned short* __restrict__ B,
                                                     void* __restrict__ Cp) {
  __shared__ __attribute__((aligned(16))) unsigned short As[2][128 * AS_STRIDE];
  __shared__ __attribute__((aligned(16))) unsigned short Bt[2][128 * BT_STRIDE];

  const int tid  = threadIdx.x;
  const int lane = tid & 31;
  const int w    = tid >> 5;
  const int half = lane >> 4;
  const int l16  = lane & 15;
  const int wy   = w >> 2;
  const int wx   = w & 3;
  const int m0   = blockIdx.y * 128;
  const int n0   = blockIdx.x * 128;

  // uniform bases + one shared per-thread offset per stream (imm deltas)
  const unsigned short* Abase = A + (size_t)m0 * K;
  const unsigned short* Bbase = B + n0;
  int ao = (tid >> 2) * K + (tid & 3) * 8;         // second chunk at +64*K (imm)
  int bo = (tid >> 6) * 2 * N + (tid & 63) * 2;    // block i at +i*8*N, pair +N

  const int sa0 = (tid >> 2) * AS_STRIDE + (tid & 3) * 8;
  const int sa1 = sa0 + 64 * AS_STRIDE;
  const int bn = (tid & 63) * 2;
  int sbl[4], sbh[4];
#pragma unroll
  for (int i = 0; i < 4; ++i) {
    int bk = ((tid >> 6) + 4 * i) * 2;
    sbl[i] = bn * BT_STRIDE + bk;
    sbh[i] = (bn + 1) * BT_STRIDE + bk;
  }

  int aoff[4], boff[2];
#pragma unroll
  for (int fy = 0; fy < 4; ++fy) aoff[fy] = (64 * wy + 16 * fy + l16) * AS_STRIDE + half * 8;
#pragma unroll
  for (int fx = 0; fx < 2; ++fx) boff[fx] = (32 * wx + 16 * fx + l16) * BT_STRIDE + half * 8;

  v8f acc[4][2];
#pragma unroll
  for (int i = 0; i < 4; ++i)
#pragma unroll
    for (int j = 0; j < 2; ++j) acc[i][j] = vzero8();

  v4u areg0, areg1;
  unsigned int breg[4][2];

  // prologue: stage tile 0
  areg0 = *(const v4u*)(Abase + ao);
  areg1 = *(const v4u*)(Abase + ao + 64 * K);
  ao += 32;
#pragma unroll
  for (int i = 0; i < 4; ++i) {
    breg[i][0] = *(const unsigned int*)(Bbase + bo + i * 8 * N);
    breg[i][1] = *(const unsigned int*)(Bbase + bo + i * 8 * N + N);
  }
  bo += 32 * N;
  *(v4u*)(As[0] + sa0) = areg0;
  *(v4u*)(As[0] + sa1) = areg1;
#pragma unroll
  for (int i = 0; i < 4; ++i) {
    unsigned int u0 = breg[i][0], u1 = breg[i][1];
    *(unsigned int*)(Bt[0] + sbl[i]) = (u0 & 0xffffu) | (u1 << 16);
    *(unsigned int*)(Bt[0] + sbh[i]) = (u0 >> 16) | (u1 & 0xffff0000u);
  }
  __syncthreads();

  constexpr int nk = K >> 5;
#pragma unroll 2
  for (int kt = 0; kt < nk; ++kt) {
    const int cur = kt & 1;
    const bool hasnext = (kt + 1) < nk;
    if (hasnext) {
      areg0 = *(const v4u*)(Abase + ao);
      areg1 = *(const v4u*)(Abase + ao + 64 * K);
      ao += 32;
#pragma unroll
      for (int i = 0; i < 4; ++i) {
        breg[i][0] = *(const unsigned int*)(Bbase + bo + i * 8 * N);
        breg[i][1] = *(const unsigned int*)(Bbase + bo + i * 8 * N + N);
      }
      bo += 32 * N;
      // keep L2 one tile ahead of the double buffer
      __builtin_prefetch(Abase + ao, 0, 2);
      __builtin_prefetch(Bbase + bo, 0, 2);
    }

    const unsigned short* as = As[cur];
    const unsigned short* bs = Bt[cur];
    FragU af[4], bf[2];
#pragma unroll
    for (int fy = 0; fy < 4; ++fy) {
      af[fy].q[0] = *(const v4u*)(as + aoff[fy]);
      af[fy].q[1] = *(const v4u*)(as + aoff[fy] + 16);
    }
#pragma unroll
    for (int fx = 0; fx < 2; ++fx) {
      bf[fx].q[0] = *(const v4u*)(bs + boff[fx]);
      bf[fx].q[1] = *(const v4u*)(bs + boff[fx] + 16);
    }
#pragma unroll
    for (int fy = 0; fy < 4; ++fy)
#pragma unroll
      for (int fx = 0; fx < 2; ++fx)
        acc[fy][fx] = wmma16(af[fy].f, bf[fx].f, acc[fy][fx]);

    if (hasnext) {
      unsigned short* asn = (unsigned short*)As[cur ^ 1];
      unsigned short* bsn = (unsigned short*)Bt[cur ^ 1];
      *(v4u*)(asn + sa0) = areg0;
      *(v4u*)(asn + sa1) = areg1;
#pragma unroll
      for (int i = 0; i < 4; ++i) {
        unsigned int u0 = breg[i][0], u1 = breg[i][1];
        *(unsigned int*)(bsn + sbl[i]) = (u0 & 0xffffu) | (u1 << 16);
        *(unsigned int*)(bsn + sbh[i]) = (u0 >> 16) | (u1 & 0xffff0000u);
      }
    }
    __syncthreads();
  }

  if (STORE_F16) {
    unsigned short* Cb = (unsigned short*)Cp + (size_t)m0 * N + n0;
#pragma unroll
    for (int fy = 0; fy < 4; ++fy)
#pragma unroll
      for (int fx = 0; fx < 2; ++fx)
#pragma unroll
        for (int r = 0; r < 8; ++r)
          Cb[(64 * wy + 16 * fy + r + half * 8) * N + 32 * wx + 16 * fx + l16] = f2h(acc[fy][fx][r]);
  } else {
    float* Cb = (float*)Cp + (size_t)m0 * N + n0;
#pragma unroll
    for (int fy = 0; fy < 4; ++fy)
#pragma unroll
      for (int fx = 0; fx < 2; ++fx)
#pragma unroll
        for (int r = 0; r < 8; ++r)
          Cb[(64 * wy + 16 * fy + r + half * 8) * N + 32 * wx + 16 * fx + l16] = acc[fy][fx][r];
  }
}

// ---------------------------------------------------------------------------
// Windowed attention (flash style). Block = (b, h, 64-query tile), 256 thr.
// ---------------------------------------------------------------------------
__global__ __launch_bounds__(256) void attn_windowed_wmma(const unsigned short* __restrict__ qb,
                                                          const unsigned short* __restrict__ kb,
                                                          const unsigned short* __restrict__ vb,
                                                          unsigned short* __restrict__ ab) {
  __shared__ __attribute__((aligned(16))) unsigned short Ks[2][64 * KS_STRIDE]; // [key][dim]
  __shared__ __attribute__((aligned(16))) unsigned short Vt[2][64 * KS_STRIDE]; // [dim][key]
  __shared__ __attribute__((aligned(16))) float Ss[64 * SS_STRIDE];             // scores (+pmax pad)
  __shared__ __attribute__((aligned(16))) unsigned short Ph[64 * KS_STRIDE];    // f16 probs (+psum pad)
  __shared__ __attribute__((aligned(16))) float mrow[64], lrow[64], arow[64];

  const int tid  = threadIdx.x;
  const int lane = tid & 31;
  const int w    = tid >> 5;
  const int half = lane >> 4;
  const int l16  = lane & 15;

  const int q0   = blockIdx.x * 64;
  const int h    = blockIdx.y;
  const int b    = blockIdx.z;
  const int kvh  = h >> 2;
  const int rowbase = 16 * (w >> 1);
  const int colsel  = (w & 1);

  const unsigned short* kB = kb + (size_t)(b * L_DIM) * KVD + kvh * HD_DIM;
  const unsigned short* vB = vb + (size_t)(b * L_DIM) * KVD + kvh * HD_DIM;

  // one shared offset per stream; block deltas are instruction immediates
  int ko = (tid >> 3) * KVD + (tid & 7) * 8;       // second chunk at +32*KVD
  const int sk0 = (tid >> 3) * KS_STRIDE + (tid & 7) * 8;
  const int sk1 = sk0 + 32 * KS_STRIDE;

  const int bdim = (tid & 31) * 2;
  int vo = (tid >> 5) * 2 * KVD + bdim;            // block i at +i*16*KVD, pair +KVD
  int sv0[4], sv1[4];
#pragma unroll
  for (int i = 0; i < 4; ++i) {
    int bkey = ((tid >> 5) + 8 * i) * 2;
    sv0[i] = bdim * KS_STRIDE + bkey;
    sv1[i] = (bdim + 1) * KS_STRIDE + bkey;
  }

  // preload Q fragments
  v16u qf[2];
  {
    const unsigned short* qB = qb + (size_t)(b * L_DIM + q0) * D_DIM + h * HD_DIM;
    int qo = (rowbase + l16) * D_DIM;
#pragma unroll
    for (int ks = 0; ks < 2; ++ks) {
      FragU fr;
      fr.q[0] = *(const v4u*)(qB + qo + 32 * ks + half * 8);
      fr.q[1] = *(const v4u*)(qB + qo + 32 * ks + 16 + half * 8);
      qf[ks] = fr.f;
    }
  }

  int kvfbase[2];
#pragma unroll
  for (int fx = 0; fx < 2; ++fx)
    kvfbase[fx] = (32 * colsel + 16 * fx + l16) * KS_STRIDE + half * 8;
  const int pbase = (rowbase + l16) * KS_STRIDE + half * 8;
  const int soff  = (rowbase + half * 8) * SS_STRIDE + 32 * colsel + l16;

  v8f of[2];
  of[0] = vzero8();
  of[1] = vzero8();

  v4u kreg0, kreg1;
  unsigned int vreg[4][2];

  // prologue: stage tile 0
  kreg0 = *(const v4u*)(kB + ko);
  kreg1 = *(const v4u*)(kB + ko + 32 * KVD);
  ko += 64 * KVD;
#pragma unroll
  for (int i = 0; i < 4; ++i) {
    vreg[i][0] = *(const unsigned int*)(vB + vo + i * 16 * KVD);
    vreg[i][1] = *(const unsigned int*)(vB + vo + i * 16 * KVD + KVD);
  }
  vo += 64 * KVD;
  *(v4u*)(Ks[0] + sk0) = kreg0;
  *(v4u*)(Ks[0] + sk1) = kreg1;
#pragma unroll
  for (int i = 0; i < 4; ++i) {
    unsigned int u0 = vreg[i][0], u1 = vreg[i][1];
    *(unsigned int*)(Vt[0] + sv0[i]) = (u0 & 0xffffu) | (u1 << 16);
    *(unsigned int*)(Vt[0] + sv1[i]) = (u0 >> 16) | (u1 & 0xffff0000u);
  }
  if (tid < 64) { mrow[tid] = -3.0e38f; lrow[tid] = 0.0f; }
  __syncthreads();

  const int row = tid & 63;   // row pass: 4 threads per row
  const int qtr = tid >> 6;   // column quarter
  const int cb  = qtr * 16;

  for (int jt = 0; jt < L_DIM / 64; ++jt) {
    const int cur = jt & 1;
    const int j0  = jt * 64;
    const bool hasnext = (jt + 1) < (L_DIM / 64);
    if (hasnext) {
      kreg0 = *(const v4u*)(kB + ko);
      kreg1 = *(const v4u*)(kB + ko + 32 * KVD);
      ko += 64 * KVD;
#pragma unroll
      for (int i = 0; i < 4; ++i) {
        vreg[i][0] = *(const unsigned int*)(vB + vo + i * 16 * KVD);
        vreg[i][1] = *(const unsigned int*)(vB + vo + i * 16 * KVD + KVD);
      }
      vo += 64 * KVD;
      __builtin_prefetch(kB + ko, 0, 2);
      __builtin_prefetch(vB + vo, 0, 2);
    }

    // ---- S = Q @ K^T ----
    {
      const unsigned short* ks = Ks[cur];
      FragU bk2[2][2];
#pragma unroll
      for (int kss = 0; kss < 2; ++kss)
#pragma unroll
        for (int fx = 0; fx < 2; ++fx) {
          bk2[kss][fx].q[0] = *(const v4u*)(ks + kvfbase[fx] + 32 * kss);
          bk2[kss][fx].q[1] = *(const v4u*)(ks + kvfbase[fx] + 32 * kss + 16);
        }
      v8f sf[2];
      sf[0] = vzero8();
      sf[1] = vzero8();
#pragma unroll
      for (int kss = 0; kss < 2; ++kss)
#pragma unroll
        for (int fx = 0; fx < 2; ++fx)
          sf[fx] = wmma16(qf[kss], bk2[kss][fx].f, sf[fx]);
#pragma unroll
      for (int fx = 0; fx < 2; ++fx)
#pragma unroll
        for (int r = 0; r < 8; ++r)
          Ss[soff + r * SS_STRIDE + 16 * fx] = sf[fx][r];
    }
    __syncthreads();

    // ---- pass1: scale + mask + partial max ----
    {
      const int lo = (q0 + row) - WIN - j0;  // masked iff (unsigned)(c-lo) < WIN
      float* srow = Ss + row * SS_STRIDE + cb;
      float lmax = -3.0e38f;
#pragma unroll
      for (int c4 = 0; c4 < 4; ++c4) {
        v4f v = *(v4f*)(srow + c4 * 4);
#pragma unroll
        for (int e = 0; e < 4; ++e) {
          int c = cb + c4 * 4 + e;
          float s = v[e] * 0.125f;
          if ((unsigned)(c - lo) < (unsigned)WIN) s += -1.0e10f;
          v[e] = s;
          lmax = fmaxf(lmax, s);
        }
        *(v4f*)(srow + c4 * 4) = v;
      }
      Ss[row * SS_STRIDE + 64 + qtr] = lmax;
    }
    __syncthreads();

    // ---- pass2: reduce max, exp + pack f16 P, partial sums ----
    {
      const float* pm = Ss + row * SS_STRIDE + 64;
      float mo = mrow[row];
      float mt = fmaxf(fmaxf(pm[0], pm[1]), fmaxf(pm[2], pm[3]));
      float mn = fmaxf(mo, mt);
      const float* srow = Ss + row * SS_STRIDE + cb;
      unsigned short* prow = Ph + row * KS_STRIDE + cb;
      float sum = 0.0f;
#pragma unroll
      for (int c8 = 0; c8 < 2; ++c8) {
        v4f u = *(const v4f*)(srow + c8 * 8);
        v4f v = *(const v4f*)(srow + c8 * 8 + 4);
        float p0 = __expf(u[0] - mn), p1 = __expf(u[1] - mn);
        float p2 = __expf(u[2] - mn), p3 = __expf(u[3] - mn);
        float p4 = __expf(v[0] - mn), p5 = __expf(v[1] - mn);
        float p6 = __expf(v[2] - mn), p7 = __expf(v[3] - mn);
        sum += ((p0 + p1) + (p2 + p3)) + ((p4 + p5) + (p6 + p7));
        v4u o;
        o[0] = pk2h(p0, p1);
        o[1] = pk2h(p2, p3);
        o[2] = pk2h(p4, p5);
        o[3] = pk2h(p6, p7);
        *(v4u*)(prow + c8 * 8) = o;
      }
      ((float*)(Ph + row * KS_STRIDE + 64))[qtr] = sum;
      if (qtr == 0) {
        arow[row] = __expf(mo - mn);
        mrow[row] = mn;
      }
    }
    __syncthreads();

    // ---- l update + O rescale + P @ V ----
    if (tid < 64) {
      const float* ps = (const float*)(Ph + tid * KS_STRIDE + 64);
      lrow[tid] = lrow[tid] * arow[tid] + (((ps[0] + ps[1]) + (ps[2] + ps[3])));
    }
    {
      v4f a0 = *(const v4f*)(arow + rowbase + half * 8);
      v4f a1 = *(const v4f*)(arow + rowbase + half * 8 + 4);
#pragma unroll
      for (int fx = 0; fx < 2; ++fx) {
#pragma unroll
        for (int r = 0; r < 4; ++r) of[fx][r] *= a0[r];
#pragma unroll
        for (int r = 4; r < 8; ++r) of[fx][r] *= a1[r - 4];
      }
    }
    {
      const unsigned short* vt = Vt[cur];
      FragU pf[2], vv[2][2];
#pragma unroll
      for (int kss = 0; kss < 2; ++kss) {
        pf[kss].q[0] = *(const v4u*)(Ph + pbase + 32 * kss);
        pf[kss].q[1] = *(const v4u*)(Ph + pbase + 32 * kss + 16);
#pragma unroll
        for (int fx = 0; fx < 2; ++fx) {
          vv[kss][fx].q[0] = *(const v4u*)(vt + kvfbase[fx] + 32 * kss);
          vv[kss][fx].q[1] = *(const v4u*)(vt + kvfbase[fx] + 32 * kss + 16);
        }
      }
#pragma unroll
      for (int kss = 0; kss < 2; ++kss)
#pragma unroll
        for (int fx = 0; fx < 2; ++fx)
          of[fx] = wmma16(pf[kss].f, vv[kss][fx].f, of[fx]);
    }

    if (hasnext) {
      unsigned short* ksn = (unsigned short*)Ks[cur ^ 1];
      unsigned short* vtn = (unsigned short*)Vt[cur ^ 1];
      *(v4u*)(ksn + sk0) = kreg0;
      *(v4u*)(ksn + sk1) = kreg1;
#pragma unroll
      for (int i = 0; i < 4; ++i) {
        unsigned int u0 = vreg[i][0], u1 = vreg[i][1];
        *(unsigned int*)(vtn + sv0[i]) = (u0 & 0xffffu) | (u1 << 16);
        *(unsigned int*)(vtn + sv1[i]) = (u0 >> 16) | (u1 & 0xffff0000u);
      }
    }
    __syncthreads();
  }

  // normalize and store f16 attn output
  {
    unsigned short* aB = ab + (size_t)(b * L_DIM + q0) * D_DIM + h * HD_DIM;
    v4f l0 = *(const v4f*)(lrow + rowbase + half * 8);
    v4f l1 = *(const v4f*)(lrow + rowbase + half * 8 + 4);
    float rinv[8];
#pragma unroll
    for (int r = 0; r < 4; ++r) rinv[r] = 1.0f / l0[r];
#pragma unroll
    for (int r = 4; r < 8; ++r) rinv[r] = 1.0f / l1[r - 4];
#pragma unroll
    for (int fx = 0; fx < 2; ++fx)
#pragma unroll
      for (int r = 0; r < 8; ++r)
        aB[(rowbase + r + half * 8) * D_DIM + 32 * colsel + 16 * fx + l16] =
            f2h(of[fx][r] * rinv[r]);
  }
}

// ---------------------------------------------------------------------------
extern "C" void kernel_launch(void* const* d_in, const int* in_sizes, int n_in,
                              void* d_out, int out_size, void* d_ws, size_t ws_size,
                              hipStream_t stream) {
  const float* x  = (const float*)d_in[0];
  const float* wq = (const float*)d_in[1];
  const float* wk = (const float*)d_in[2];
  const float* wv = (const float*)d_in[3];
  const float* wo = (const float*)d_in[4];
  float* out = (float*)d_out;

  const int nX  = B_DIM * L_DIM * D_DIM;
  const int nWQ = D_DIM * D_DIM;
  const int nWK = D_DIM * KVD;
  const int nK  = B_DIM * L_DIM * KVD;

  char* ws = (char*)d_ws;
  size_t off = 0;
  unsigned short* xh  = (unsigned short*)(ws + off); off += (size_t)nX  * 2;
  unsigned short* wqh = (unsigned short*)(ws + off); off += (size_t)nWQ * 2;
  unsigned short* wkh = (unsigned short*)(ws + off); off += (size_t)nWK * 2;
  unsigned short* wvh = (unsigned short*)(ws + off); off += (size_t)nWK * 2;
  unsigned short* woh = (unsigned short*)(ws + off); off += (size_t)nWQ * 2;
  unsigned short* qh  = (unsigned short*)(ws + off); off += (size_t)nX  * 2;
  unsigned short* kh  = (unsigned short*)(ws + off); off += (size_t)nK  * 2;
  unsigned short* vh  = (unsigned short*)(ws + off); off += (size_t)nK  * 2;
  unsigned short* ah  = (unsigned short*)(ws + off); off += (size_t)nX  * 2;

  cvt_f32_to_f16<<<(nX  / 8 + 255) / 256, 256, 0, stream>>>(x,  xh,  nX  / 8);
  cvt_f32_to_f16<<<(nWQ / 8 + 255) / 256, 256, 0, stream>>>(wq, wqh, nWQ / 8);
  cvt_f32_to_f16<<<(nWK / 8 + 255) / 256, 256, 0, stream>>>(wk, wkh, nWK / 8);
  cvt_f32_to_f16<<<(nWK / 8 + 255) / 256, 256, 0, stream>>>(wv, wvh, nWK / 8);
  cvt_f32_to_f16<<<(nWQ / 8 + 255) / 256, 256, 0, stream>>>(wo, woh, nWQ / 8);

  // projections (M = 4096 rows)
  gemm_f16_wmma<true, D_DIM, D_DIM><<<dim3(D_DIM / 128, 32), 256, 0, stream>>>(xh, wqh, (void*)qh);
  gemm_f16_wmma<true, KVD,   D_DIM><<<dim3(KVD   / 128, 32), 256, 0, stream>>>(xh, wkh, (void*)kh);
  gemm_f16_wmma<true, KVD,   D_DIM><<<dim3(KVD   / 128, 32), 256, 0, stream>>>(xh, wvh, (void*)vh);

  attn_windowed_wmma<<<dim3(L_DIM / 64, H_DIM, B_DIM), 256, 0, stream>>>(qh, kh, vh, ah);

  gemm_f16_wmma<false, D_DIM, D_DIM><<<dim3(D_DIM / 128, 32), 256, 0, stream>>>(ah, woh, (void*)out);
}
